// ComplexS4Model_24343874633775
// MI455X (gfx1250) — compile-verified, matrix operand order checked by reading, not run on hardware
//
#include <hip/hip_runtime.h>
#include <hip/hip_bf16.h>
#include <stddef.h>

// ---------------- problem dims (match reference) ----------------
#define BB   8
#define LL   4096
#define DIN  256
#define HH   256
#define DOUT 256
#define NL   4
#define NN   64
#define BL   (BB * LL)          // 32768 tokens
#define CHUNK 512               // u-columns staged per LDS round

typedef __bf16 bf16;
typedef __attribute__((ext_vector_type(16))) __bf16 v16bf;
typedef __attribute__((ext_vector_type(8)))  float  v8f;
typedef __attribute__((ext_vector_type(4)))  int    v4i;

#define WMMA_BF16(A, Bm, C) \
  __builtin_amdgcn_wmma_f32_16x16x32_bf16(false, (A), false, (Bm), (short)0, (C), false, false)

// Documented 16-bit A-matrix 16x32 layout: lanes 0-15 hold K 0..7 (slots 0..7)
// and K 16..23 (slots 8..15); lanes 16-31 hold K 8..15 and 24..31.
__device__ __forceinline__ int frag_k(int e, int half) {
  return (e & 7) + ((e >> 3) << 4) + (half << 3);
}

__device__ __forceinline__ float gelu_tanh(float y) {
  float t = 0.7978845608028654f * (y + 0.044715f * y * y * y);
  return 0.5f * y * (1.0f + tanhf(t));
}

// ---- gfx1250 async global->LDS copy (16 bytes), ASYNCcnt-tracked ----
// Builtin signature (from compiler diagnostic): (int4 AS1*, int4 AS3*, imm, imm)
typedef __attribute__((address_space(1))) v4i g_v4i;
typedef __attribute__((address_space(3))) v4i l_v4i;

__device__ __forceinline__ void async_copy16(const void* g, void* l) {
#if __has_builtin(__builtin_amdgcn_global_load_async_to_lds_b128)
  // generic->AS1: same 64-bit VA; generic LDS ptr low 32 bits == AS3 offset.
  __builtin_amdgcn_global_load_async_to_lds_b128(
      (g_v4i*)(unsigned long long)g,
      (l_v4i*)(unsigned int)(unsigned long long)l, 0, 0);
#else
  *(uint4*)l = *(const uint4*)g;   // sync fallback: vmem load + ds store
#endif
}

__device__ __forceinline__ void wait_async_barrier() {
#if __has_builtin(__builtin_amdgcn_s_wait_asynccnt)
  __builtin_amdgcn_s_wait_asynccnt(0);
#elif defined(__gfx1250__)
  asm volatile("s_wait_asynccnt 0" ::: "memory");
#endif
  __syncthreads();
}

// ---------------- f32 -> bf16 convert ----------------
__global__ void cvt_bf16(const float* __restrict__ src, bf16* __restrict__ dst, int n) {
  int i = blockIdx.x * blockDim.x + threadIdx.x;
  if (i < n) dst[i] = (bf16)src[i];
}

// ---------------- encoder: complex linear DIN->H, WMMA bf16 ----------------
// Block = 8 waves sharing one 16-row h-tile; W rows staged via async->LDS.
// out: xr/xi f32 (B,H,L) and u16 bf16 [h][c16][l], c16 = b*2 + (re/im)
__global__ void enc_gemm(const float* __restrict__ x,
                         const bf16* __restrict__ Wr, const bf16* __restrict__ Wi,
                         const float* __restrict__ br, const float* __restrict__ bi,
                         float* __restrict__ xr, float* __restrict__ xi,
                         bf16* __restrict__ u16) {
  __shared__ __align__(16) bf16 Wl[2][16][DIN];       // 16 KB
  const int tid  = threadIdx.x;
  const int lane = tid & 31;
  const int wid  = blockIdx.x * (blockDim.x >> 5) + (tid >> 5);
  const int mtiles = BL / 16;                 // 2048 (multiple of 8 -> h0 uniform/block)
  const int h0 = (wid / mtiles) * 16;
  const int m0 = (wid % mtiles) * 16;
  const int half = lane >> 4, mn = lane & 15;

  // stage Wr/Wi rows [h0, h0+16) : 2 * 16 * 256 bf16 = 8192 elements
  for (int p = tid * 8; p < 16 * DIN; p += 256 * 8) {
    const int row = p / DIN, col = p % DIN;
    async_copy16(&Wr[(h0 + row) * DIN + col], &Wl[0][row][col]);
    async_copy16(&Wi[(h0 + row) * DIN + col], &Wl[1][row][col]);
  }
  wait_async_barrier();

  v8f arr = {}, aii = {}, ari = {}, air = {};
  for (int k0 = 0; k0 < DIN; k0 += 32) {
    v16bf awr, awi, bxr, bxi;
    const int bm = m0 + mn;                   // B column = token index
#pragma unroll
    for (int e = 0; e < 16; ++e) {
      const int k = k0 + frag_k(e, half);
      awr[e] = Wl[0][mn][k];
      awi[e] = Wl[1][mn][k];
      bxr[e] = (bf16)x[((size_t)bm * DIN + k) * 2 + 0];
      bxi[e] = (bf16)x[((size_t)bm * DIN + k) * 2 + 1];
    }
    arr = WMMA_BF16(awr, bxr, arr);
    aii = WMMA_BF16(awi, bxi, aii);
    ari = WMMA_BF16(awr, bxi, ari);
    air = WMMA_BF16(awi, bxr, air);
  }
  const int m = m0 + mn;                      // token = b*L + l
  const int b = m / LL, l = m % LL;
#pragma unroll
  for (int r = 0; r < 8; ++r) {
    const int h = h0 + r + 8 * half;
    const float yr = arr[r] - aii[r] + br[h];
    const float yi = ari[r] + air[r] + bi[h];
    xr[((size_t)b * HH + h) * LL + l] = yr;
    xi[((size_t)b * HH + h) * LL + l] = yi;
    u16[((size_t)h * 16 + b * 2 + 0) * LL + l] = (bf16)yr;
    u16[((size_t)h * 16 + b * 2 + 1) * LL + l] = (bf16)yi;
  }
}

// ---------------- S4D Vandermonde kernel K[h,l] (bf16) ----------------
__global__ void kgen(const float* __restrict__ log_dt, const float* __restrict__ A_re_log,
                     const float* __restrict__ A_im,   const float* __restrict__ C_re,
                     const float* __restrict__ C_im,   int layer, bf16* __restrict__ Kbf) {
  const int idx = blockIdx.x * blockDim.x + threadIdx.x;
  if (idx >= HH * LL) return;
  const int h = idx / LL, l = idx % LL;
  const float dt = __expf(log_dt[layer * HH + h]);
  float acc = 0.f;
  for (int n = 0; n < NN; ++n) {
    const int p = (layer * HH + h) * NN + n;
    const float Are = -__expf(A_re_log[p]);
    const float Aim = A_im[p];
    const float dAr = Are * dt, dAi = Aim * dt;
    const float er = __expf(dAr);
    float s, c; __sincosf(dAi, &s, &c);
    const float e1r = er * c - 1.f, e1i = er * s;          // expm1(dt*A)
    const float den = Are * Are + Aim * Aim;
    const float qr = (e1r * Are + e1i * Aim) / den;        // expm1(dt*A)/A
    const float qi = (e1i * Are - e1r * Aim) / den;
    const float Cr = C_re[p], Ci = C_im[p];
    const float Ctr = Cr * qr - Ci * qi, Cti = Cr * qi + Ci * qr;
    const float elr = __expf(dAr * (float)l);              // exp(dt*A*l)
    float sl, cl; __sincosf(dAi * (float)l, &sl, &cl);
    acc += Ctr * (elr * cl) - Cti * (elr * sl);
  }
  Kbf[idx] = (bf16)(2.f * acc);
}

// ---------------- causal Toeplitz conv via WMMA, LDS-staged ----------------
// One block = channel h, 128 output rows (8 waves x 16).  K prefix and u
// chunks are async-DMA'd into LDS; fragments are built from LDS.
// g16 out: [c16][l][h] so GLU GEMM B-fragments read contiguous h.
__global__ void ssm_toeplitz(const bf16* __restrict__ Kbf, const bf16* __restrict__ u16,
                             const float* __restrict__ xr, const float* __restrict__ xi,
                             const float* __restrict__ Dvec, int layer,
                             bf16* __restrict__ g16) {
  __shared__ __align__(16) bf16 Kbuf[LL];           // 8 KB (prefix [0, mb0+128))
  __shared__ __align__(16) bf16 Ubuf[16][CHUNK];    // 16 KB
  const int tid  = threadIdx.x;
  const int lane = tid & 31, wave = tid >> 5;
  const int strips = LL / 128;                      // 32 strips per h
  const int h   = blockIdx.x / strips;
  const int mb0 = (blockIdx.x % strips) * 128;
  const int m0  = mb0 + wave * 16;
  const int half = lane >> 4, mn = lane & 15;

  const bf16* __restrict__ Kh = Kbf + (size_t)h * LL;
  const bf16* __restrict__ Uh = u16 + (size_t)h * 16 * LL;

  // stage K prefix [0, mb0+128) -- multiple of 128 elements
  const int klen = mb0 + 128;
  for (int p = tid * 8; p < klen; p += 256 * 8)
    async_copy16(&Kh[p], &Kbuf[p]);

  v8f acc = {};
  for (int j0 = 0; j0 <= mb0; j0 += CHUNK) {
    // stage u chunk: 16 streams x CHUNK columns = 8192 bf16
    for (int p = tid * 8; p < 16 * CHUNK; p += 256 * 8) {
      const int cc = p / CHUNK, ll = p % CHUNK;
      async_copy16(&Uh[(size_t)cc * LL + j0 + ll], &Ubuf[cc][ll]);
    }
    wait_async_barrier();

    const int jmax = min(CHUNK, m0 + 16 - j0);      // causal extent for this wave
    for (int jj = 0; jj < jmax; jj += 32) {
      v16bf a, bm;
#pragma unroll
      for (int e = 0; e < 16; ++e) {
        const int kk = frag_k(e, half);
        const int d  = (m0 + mn) - (j0 + jj + kk);  // Toeplitz K index
        a[e]  = (d >= 0) ? Kbuf[d] : (bf16)0.0f;
        bm[e] = Ubuf[mn][jj + kk];                  // column = stream mn
      }
      acc = WMMA_BF16(a, bm, acc);
    }
    __syncthreads();                                // protect Ubuf before next stage
  }

  const int c = mn, b = c >> 1, ri = c & 1;
  const float Dh = Dvec[layer * HH + h];
  const float* __restrict__ xs = ri ? xi : xr;
#pragma unroll
  for (int r = 0; r < 8; ++r) {
    const int l = m0 + r + 8 * half;
    const float y = acc[r] + Dh * xs[((size_t)b * HH + h) * LL + l];
    g16[((size_t)c * LL + l) * HH + h] = (bf16)gelu_tanh(y);
  }
}

// ---------------- GLU projection: v = W(2H x H) @ g, z = v1 * sigmoid(v2) ----------------
// Block shares one (g0, stream) pair: W rows + gate rows staged via async->LDS.
__global__ void glu_gemm(const bf16* __restrict__ Wg, const float* __restrict__ out_b,
                         int layer, const bf16* __restrict__ g16,
                         float* __restrict__ xr, float* __restrict__ xi) {
  __shared__ __align__(16) bf16 Wl[2][16][HH];      // 16 KB
  const int tid  = threadIdx.x;
  const int lane = tid & 31;
  const int wid  = blockIdx.x * (blockDim.x >> 5) + (tid >> 5);
  const int ltiles = LL / 16, gtiles = HH / 16;     // 256, 16 (256 % 8 == 0)
  int t = wid;
  const int lt = t % ltiles; t /= ltiles;
  const int gt = t % gtiles; t /= gtiles;
  const int c  = t;                                 // stream 0..15 (uniform per block)
  const int g0 = gt * 16, l0 = lt * 16;
  const int half = lane >> 4, mn = lane & 15;
  const bf16* __restrict__ gc = g16 + (size_t)c * LL * HH;

  for (int p = tid * 8; p < 16 * HH; p += 256 * 8) {
    const int row = p / HH, col = p % HH;
    async_copy16(&Wg[(g0 + row) * HH + col],      &Wl[0][row][col]);
    async_copy16(&Wg[(HH + g0 + row) * HH + col], &Wl[1][row][col]);
  }
  wait_async_barrier();

  v8f acc1 = {}, acc2 = {};
  for (int k0 = 0; k0 < HH; k0 += 32) {
    v16bf a1, a2, bm;
#pragma unroll
    for (int e = 0; e < 16; ++e) {
      const int k = k0 + frag_k(e, half);
      a1[e] = Wl[0][mn][k];
      a2[e] = Wl[1][mn][k];
      bm[e] = gc[(size_t)(l0 + mn) * HH + k];
    }
    acc1 = WMMA_BF16(a1, bm, acc1);
    acc2 = WMMA_BF16(a2, bm, acc2);
  }
  const int b = c >> 1, ri = c & 1;
  float* __restrict__ xs = ri ? xi : xr;
  const float* __restrict__ ob = out_b + layer * 2 * HH;
  const int l = l0 + mn;
#pragma unroll
  for (int r = 0; r < 8; ++r) {
    const int h = g0 + r + 8 * half;
    const float v1 = acc1[r] + ob[h];
    const float v2 = acc2[r] + ob[HH + h];
    const float z  = v1 * (1.0f / (1.0f + __expf(-v2)));
    xs[((size_t)b * HH + h) * LL + l] += z;          // residual
  }
}

// ---------------- channel LayerNorm (post-norm) ----------------
__global__ void ln_kernel(float* __restrict__ xr, float* __restrict__ xi,
                          const float* __restrict__ ln_wr, const float* __restrict__ ln_br,
                          const float* __restrict__ ln_wi, const float* __restrict__ ln_bi,
                          int layer, bf16* __restrict__ u16_next,
                          bf16* __restrict__ xT16, int is_last) {
  const int idx = blockIdx.x * blockDim.x + threadIdx.x;
  if (idx >= BL * 2) return;
  const int ri = idx & 1;
  const int m  = idx >> 1;
  const int b  = m / LL, l = m % LL;
  float* __restrict__ xs = ri ? xi : xr;
  const float* __restrict__ w  = (ri ? ln_wi : ln_wr) + layer * HH;
  const float* __restrict__ bb = (ri ? ln_bi : ln_br) + layer * HH;
  float mean = 0.f, sq = 0.f;
  for (int h = 0; h < HH; ++h) {
    const float v = xs[((size_t)b * HH + h) * LL + l];
    mean += v; sq += v * v;
  }
  mean *= (1.0f / HH);
  const float var = sq * (1.0f / HH) - mean * mean;
  const float inv = rsqrtf(var + 1e-5f);
  for (int h = 0; h < HH; ++h) {
    const size_t p = ((size_t)b * HH + h) * LL + l;
    const float v = (xs[p] - mean) * inv * w[h] + bb[h];
    xs[p] = v;
    if (!is_last) u16_next[((size_t)h * 16 + b * 2 + ri) * LL + l] = (bf16)v;
    else          xT16[((size_t)ri * BL + m) * HH + h] = (bf16)v;
  }
}

// ---------------- decoder: complex linear H->DOUT, WMMA bf16 ----------------
__global__ void dec_gemm(const bf16* __restrict__ xT,
                         const bf16* __restrict__ Wr, const bf16* __restrict__ Wi,
                         const float* __restrict__ br, const float* __restrict__ bi,
                         float* __restrict__ out) {
  __shared__ __align__(16) bf16 Wl[2][16][HH];      // 16 KB
  const int tid  = threadIdx.x;
  const int lane = tid & 31;
  const int wid  = blockIdx.x * (blockDim.x >> 5) + (tid >> 5);
  const int mtiles = BL / 16;
  const int o0 = (wid / mtiles) * 16;
  const int m0 = (wid % mtiles) * 16;
  const int half = lane >> 4, mn = lane & 15;
  const bf16* __restrict__ xTr = xT;
  const bf16* __restrict__ xTi = xT + (size_t)BL * HH;

  for (int p = tid * 8; p < 16 * HH; p += 256 * 8) {
    const int row = p / HH, col = p % HH;
    async_copy16(&Wr[(o0 + row) * HH + col], &Wl[0][row][col]);
    async_copy16(&Wi[(o0 + row) * HH + col], &Wl[1][row][col]);
  }
  wait_async_barrier();

  v8f arr = {}, aii = {}, ari = {}, air = {};
  for (int k0 = 0; k0 < HH; k0 += 32) {
    v16bf awr, awi, bxr, bxi;
    const int bm = m0 + mn;
#pragma unroll
    for (int e = 0; e < 16; ++e) {
      const int k = k0 + frag_k(e, half);
      awr[e] = Wl[0][mn][k];
      awi[e] = Wl[1][mn][k];
      bxr[e] = xTr[(size_t)bm * HH + k];
      bxi[e] = xTi[(size_t)bm * HH + k];
    }
    arr = WMMA_BF16(awr, bxr, arr);
    aii = WMMA_BF16(awi, bxi, aii);
    ari = WMMA_BF16(awr, bxi, ari);
    air = WMMA_BF16(awi, bxr, air);
  }
  const int m = m0 + mn;
#pragma unroll
  for (int r = 0; r < 8; ++r) {
    const int o = o0 + r + 8 * half;
    out[((size_t)m * DOUT + o) * 2 + 0] = arr[r] - aii[r] + br[o];
    out[((size_t)m * DOUT + o) * 2 + 1] = ari[r] + air[r] + bi[o];
  }
}

// ---------------- host orchestration ----------------
extern "C" void kernel_launch(void* const* d_in, const int* in_sizes, int n_in,
                              void* d_out, int out_size, void* d_ws, size_t ws_size,
                              hipStream_t stream) {
  const float* x        = (const float*)d_in[0];
  const float* enc_Wr   = (const float*)d_in[1];
  const float* enc_Wi   = (const float*)d_in[2];
  const float* enc_br   = (const float*)d_in[3];
  const float* enc_bi   = (const float*)d_in[4];
  const float* log_dt   = (const float*)d_in[5];
  const float* A_re_log = (const float*)d_in[6];
  const float* A_im     = (const float*)d_in[7];
  const float* C_re     = (const float*)d_in[8];
  const float* C_im     = (const float*)d_in[9];
  const float* Dvec     = (const float*)d_in[10];
  const float* out_W    = (const float*)d_in[11];
  const float* out_b    = (const float*)d_in[12];
  const float* ln_wr    = (const float*)d_in[13];
  const float* ln_br    = (const float*)d_in[14];
  const float* ln_wi    = (const float*)d_in[15];
  const float* ln_bi    = (const float*)d_in[16];
  const float* dec_Wr   = (const float*)d_in[17];
  const float* dec_Wi   = (const float*)d_in[18];
  const float* dec_br   = (const float*)d_in[19];
  const float* dec_bi   = (const float*)d_in[20];
  float* out = (float*)d_out;

  // workspace carve-up
  char* ws = (char*)d_ws;
  size_t off = 0;
  float* xr   = (float*)(ws + off); off += (size_t)BB * HH * LL * 4;         // 32 MB
  float* xi   = (float*)(ws + off); off += (size_t)BB * HH * LL * 4;         // 32 MB
  bf16*  u16  = (bf16*)(ws + off);  off += (size_t)HH * 16 * LL * 2;         // 32 MB
  bf16*  g16  = (bf16*)(ws + off);  off += (size_t)16 * LL * HH * 2;         // 32 MB
  bf16*  Kbf  = (bf16*)(ws + off);  off += (size_t)HH * LL * 2;              //  2 MB
  bf16*  xT16 = (bf16*)(ws + off);  off += (size_t)2 * BL * HH * 2;          // 32 MB
  bf16*  Wenc_r = (bf16*)(ws + off); off += (size_t)HH * DIN * 2;
  bf16*  Wenc_i = (bf16*)(ws + off); off += (size_t)HH * DIN * 2;
  bf16*  Wout   = (bf16*)(ws + off); off += (size_t)2 * HH * HH * 2;
  bf16*  Wdec_r = (bf16*)(ws + off); off += (size_t)DOUT * HH * 2;
  bf16*  Wdec_i = (bf16*)(ws + off); off += (size_t)DOUT * HH * 2;
  (void)ws_size; (void)n_in; (void)in_sizes; (void)out_size;

  const int T = 256;  // 8 waves / workgroup

  // weight conversions
  cvt_bf16<<<(HH * DIN + T - 1) / T, T, 0, stream>>>(enc_Wr, Wenc_r, HH * DIN);
  cvt_bf16<<<(HH * DIN + T - 1) / T, T, 0, stream>>>(enc_Wi, Wenc_i, HH * DIN);
  cvt_bf16<<<(DOUT * HH + T - 1) / T, T, 0, stream>>>(dec_Wr, Wdec_r, DOUT * HH);
  cvt_bf16<<<(DOUT * HH + T - 1) / T, T, 0, stream>>>(dec_Wi, Wdec_i, DOUT * HH);

  // encoder: (H/16)*(BL/16) = 32768 waves -> 4096 blocks
  enc_gemm<<<(HH / 16) * (BL / 16) / 8, T, 0, stream>>>(
      x, Wenc_r, Wenc_i, enc_br, enc_bi, xr, xi, u16);

  for (int i = 0; i < NL; ++i) {
    kgen<<<(HH * LL + T - 1) / T, T, 0, stream>>>(log_dt, A_re_log, A_im, C_re, C_im, i, Kbf);
    cvt_bf16<<<(2 * HH * HH + T - 1) / T, T, 0, stream>>>(out_W + (size_t)i * 2 * HH * HH,
                                                          Wout, 2 * HH * HH);
    // one block per (h, 128-row strip): 256 * 32 = 8192 blocks
    ssm_toeplitz<<<HH * (LL / 128), T, 0, stream>>>(Kbf, u16, xr, xi, Dvec, i, g16);
    // 16 streams * 16 gtiles * 256 ltiles = 65536 waves -> 8192 blocks
    glu_gemm<<<16 * (HH / 16) * (LL / 16) / 8, T, 0, stream>>>(Wout, out_b, i, g16, xr, xi);
    ln_kernel<<<(BL * 2 + T - 1) / T, T, 0, stream>>>(
        xr, xi, ln_wr, ln_br, ln_wi, ln_bi, i, u16, xT16, (i == NL - 1) ? 1 : 0);
  }

  // decoder: 32768 waves -> 4096 blocks
  dec_gemm<<<(DOUT / 16) * (BL / 16) / 8, T, 0, stream>>>(
      xT16, Wdec_r, Wdec_i, dec_br, dec_bi, out);
}
